// DCN_14688788152408
// MI455X (gfx1250) — compile-verified, matrix operand order checked by reading, not run on hardware
//
#include <hip/hip_runtime.h>
#include <hip/hip_bf16.h>

typedef _Float16 v16h __attribute__((ext_vector_type(16)));
typedef _Float16 v8h  __attribute__((ext_vector_type(8)));
typedef float    v8f  __attribute__((ext_vector_type(8)));
typedef float    v4f  __attribute__((ext_vector_type(4)));

#define Bsz 8
#define Tsz 256
#define Nsz 207
#define Csz 128
#define DIL 2
#define LN_EPS 1e-5f

#define HSTRIDE 136   // 16x128 f16 h-tile, padded (272B rows -> conflict-free b128 reads)
#define GSTRIDE 132   // 16x128 f32 gated tile, padded

#define LOG2E 1.4426950408889634f

// Branchless erf (Abramowitz-Stegun 7.1.26, |err| <= 1.5e-7), native exp2/rcp.
__device__ __forceinline__ float erf_fast(float x) {
    const float a = __builtin_fabsf(x);
    const float t = __builtin_amdgcn_rcpf(__builtin_fmaf(0.3275911f, a, 1.0f));
    float p;
    p = __builtin_fmaf(1.061405429f, t, -1.453152027f);
    p = __builtin_fmaf(p, t,  1.421413741f);
    p = __builtin_fmaf(p, t, -0.284496736f);
    p = __builtin_fmaf(p, t,  0.254829592f);
    p = p * t;
    const float e = __builtin_amdgcn_exp2f(-a * a * LOG2E);  // e^{-x^2}
    const float r = __builtin_fmaf(-p, e, 1.0f);
    return __builtin_copysignf(r, x);
}

__device__ __forceinline__ float gelu_fast(float z) {
    return 0.5f * z * (1.0f + erf_fast(z * 0.70710678118654752f));
}

__device__ __forceinline__ float sigmoid_fast(float u) {
    const float e = __builtin_amdgcn_exp2f(-u * LOG2E);      // e^{-u}
    return __builtin_amdgcn_rcpf(1.0f + e);
}

__global__ __launch_bounds__(256, 1)
void dcn_fused_kernel(const float* __restrict__ x,
                      const float* __restrict__ dw_w,
                      const float* __restrict__ dw_b,
                      const float* __restrict__ fw_w,
                      const float* __restrict__ fw_b,
                      const float* __restrict__ ln_g,
                      const float* __restrict__ ln_b,
                      float* __restrict__ out)
{
    __shared__ _Float16 hlds[16 * HSTRIDE];
    __shared__ float    glds[16 * GSTRIDE];

    const int tid  = threadIdx.x;
    const int wave = tid >> 5;
    const int lane = tid & 31;
    const int ln16 = lane & 15;
    const int hi16 = lane >> 4;

    // ------------------------------------------------------------------
    // B-matrix fragments (pointwise weights), f32 -> f16, held in VGPRs.
    // 16-bit operand striping: lanes 0-15 hold K {0..7,16..23} of chunk,
    // lanes 16-31 hold K {8..15,24..31}.  Row (= output channel) = lane&15.
    // ------------------------------------------------------------------
    v16h bf[4], bg[4];
    {
        const int koff = hi16 * 8;
        const int of = 16 * wave + ln16;        // z_f channel
        const int og = 128 + 16 * wave + ln16;  // z_g channel
#pragma unroll
        for (int q = 0; q < 4; ++q) {
            const float* pf = fw_w + (size_t)of * Csz + 32 * q + koff;
            const float* pg = fw_w + (size_t)og * Csz + 32 * q + koff;
            v4f f0 = *(const v4f*)(pf);      v4f f1 = *(const v4f*)(pf + 4);
            v4f f2 = *(const v4f*)(pf + 16); v4f f3 = *(const v4f*)(pf + 20);
            v4f g0 = *(const v4f*)(pg);      v4f g1 = *(const v4f*)(pg + 4);
            v4f g2 = *(const v4f*)(pg + 16); v4f g3 = *(const v4f*)(pg + 20);
#pragma unroll
            for (int i = 0; i < 4; ++i) {
                bf[q][i]      = (_Float16)f0[i];
                bf[q][4 + i]  = (_Float16)f1[i];
                bf[q][8 + i]  = (_Float16)f2[i];
                bf[q][12 + i] = (_Float16)f3[i];
                bg[q][i]      = (_Float16)g0[i];
                bg[q][4 + i]  = (_Float16)g1[i];
                bg[q][8 + i]  = (_Float16)g2[i];
                bg[q][12 + i] = (_Float16)g3[i];
            }
        }
    }
    const float biasF = fw_b[16 * wave + ln16];
    const float biasG = fw_b[128 + 16 * wave + ln16];

    // per-thread channel slice for phases 1 & 3
    const int r  = tid >> 4;          // row 0..15 within tile
    const int c0 = (tid & 15) * 8;    // channel base
    float w0[8], w1[8], w2[8], db[8], lg[8], lb[8];
#pragma unroll
    for (int j = 0; j < 8; ++j) {
        const int c = c0 + j;
        w0[j] = dw_w[c * 3 + 0];   // tap at t-4
        w1[j] = dw_w[c * 3 + 1];   // tap at t-2
        w2[j] = dw_w[c * 3 + 2];   // tap at t
        db[j] = dw_b[c];
        lg[j] = ln_g[c];
        lb[j] = ln_b[c];
    }

    for (int mt = 0; mt < 4; ++mt) {
        const int pbase = blockIdx.x * 64 + mt * 16;

        // -------- phase 1: causal dilated depthwise conv -> f16 LDS tile
        {
            const int p  = pbase + r;
            const int bt = p / Nsz;              // b*T + t
            const int t  = bt & (Tsz - 1);       // T = 256 (pow2)
            const float* x0 = x + (size_t)p * Csz + c0;
            float h[8];
            {
                v4f a = *(const v4f*)(x0);
                v4f b = *(const v4f*)(x0 + 4);
#pragma unroll
                for (int j = 0; j < 4; ++j) {
                    h[j]     = db[j]     + w2[j]     * a[j];
                    h[4 + j] = db[4 + j] + w2[4 + j] * b[j];
                }
            }
            if (t >= DIL) {
                const float* xm = x0 - (size_t)DIL * Nsz * Csz;
                v4f a = *(const v4f*)(xm);
                v4f b = *(const v4f*)(xm + 4);
#pragma unroll
                for (int j = 0; j < 4; ++j) {
                    h[j]     += w1[j]     * a[j];
                    h[4 + j] += w1[4 + j] * b[j];
                }
            }
            if (t >= 2 * DIL) {
                const float* xm = x0 - (size_t)(2 * DIL) * Nsz * Csz;
                v4f a = *(const v4f*)(xm);
                v4f b = *(const v4f*)(xm + 4);
#pragma unroll
                for (int j = 0; j < 4; ++j) {
                    h[j]     += w0[j]     * a[j];
                    h[4 + j] += w0[4 + j] * b[j];
                }
            }
            v8h hp;
#pragma unroll
            for (int j = 0; j < 8; ++j) hp[j] = (_Float16)h[j];
            *(v8h*)(hlds + r * HSTRIDE + c0) = hp;
        }
        __syncthreads();

        // -------- phase 2: WMMA GEMM (z = h @ W^T) + branchless GELU*sigmoid
        {
            v8f cf = {}, cg = {};
#pragma unroll
            for (int q = 0; q < 4; ++q) {
                const int base = ln16 * HSTRIDE + 32 * q + hi16 * 8;
                v8h lo = *(const v8h*)(hlds + base);
                v8h hh = *(const v8h*)(hlds + base + 16);
                v16h a;
#pragma unroll
                for (int i = 0; i < 8; ++i) { a[i] = lo[i]; a[i + 8] = hh[i]; }
                cf = __builtin_amdgcn_wmma_f32_16x16x32_f16(
                        false, a, false, bf[q], (short)0, cf, false, false);
                cg = __builtin_amdgcn_wmma_f32_16x16x32_f16(
                        false, a, false, bg[q], (short)0, cg, false, false);
            }
            const int col = 16 * wave + ln16;
#pragma unroll
            for (int v = 0; v < 8; ++v) {
                const float zf = cf[v] + biasF;
                const float zg = cg[v] + biasG;
                glds[(v + hi16 * 8) * GSTRIDE + col] =
                    gelu_fast(zf) * sigmoid_fast(zg);
            }
        }
        __syncthreads();

        // -------- phase 3: LayerNorm over 128 channels + coalesced store
        {
            const int p = pbase + r;
            const float* gp = glds + r * GSTRIDE + c0;
            float g[8];
            float s = 0.f, ss = 0.f;
#pragma unroll
            for (int j = 0; j < 8; ++j) {
                g[j] = gp[j];
                s  += g[j];
                ss += g[j] * g[j];
            }
            // 16 threads per row live in one 16-lane half of a wave32
#pragma unroll
            for (int m = 8; m >= 1; m >>= 1) {
                s  += __shfl_xor(s,  m, 32);
                ss += __shfl_xor(ss, m, 32);
            }
            const float mu   = s * (1.0f / 128.0f);
            const float var  = ss * (1.0f / 128.0f) - mu * mu;
            const float rsig = rsqrtf(var + LN_EPS);
            v4f o0, o1;
#pragma unroll
            for (int j = 0; j < 4; ++j) {
                o0[j] = (g[j]     - mu) * rsig * lg[j]     + lb[j];
                o1[j] = (g[4 + j] - mu) * rsig * lg[4 + j] + lb[4 + j];
            }
            float* op = out + (size_t)p * Csz + c0;
            *(v4f*)(op)     = o0;
            *(v4f*)(op + 4) = o1;
        }
        __syncthreads();
    }
}

extern "C" void kernel_launch(void* const* d_in, const int* in_sizes, int n_in,
                              void* d_out, int out_size, void* d_ws, size_t ws_size,
                              hipStream_t stream) {
    const float* x    = (const float*)d_in[0];
    const float* dw_w = (const float*)d_in[1];
    const float* dw_b = (const float*)d_in[2];
    const float* fw_w = (const float*)d_in[3];
    const float* fw_b = (const float*)d_in[4];
    const float* ln_g = (const float*)d_in[5];
    const float* ln_b = (const float*)d_in[6];
    // d_in[7] is the dilation scalar (== 2); hardcoded as DIL.

    const int M      = Bsz * Tsz * Nsz;  // 423,936 positions
    const int blocks = M / 64;           // 6,624 blocks x 64 rows, exact
    dcn_fused_kernel<<<blocks, 256, 0, stream>>>(x, dw_w, dw_b, fw_w, fw_b,
                                                 ln_g, ln_b, (float*)d_out);
}
